// GeometryOptimalTransport_13889924235602
// MI455X (gfx1250) — matrix operand order repaired and checked by reading, not compiled, not run on hardware
//
#include <hip/hip_runtime.h>
#include <hip/hip_bf16.h>
#include <stdint.h>

// GeometryOptimalTransport for MI455X (gfx1250, wave32).
// Implicit 4x4096x4096 cost matrix recomputed from LDS-resident 2-D locations
// (never materialized in HBM); Sinkhorn u/v via streaming stabilized
// logsumexp; final einsum fused with attn generation on
// v_wmma_f32_16x16x32_f16 with double-buffered f16 B tiles in LDS.

#define B_  4
#define T_  4096
#define S_  4096
#define D_  64
#define T2F 0.04f          // DIST_THRESH^2
#define BIGNEG (-1.0e30f)  // masked entries: exp underflows to exactly 0
#define NITER 8

typedef _Float16 v16h __attribute__((ext_vector_type(16)));
typedef _Float16 v8h  __attribute__((ext_vector_type(8)));
typedef float    v8f  __attribute__((ext_vector_type(8)));

__device__ __forceinline__ float einv() { return 1.0f / (0.01f + 1e-8f); }

__global__ void init_vec(float* v, int n) {
  int i = blockIdx.x * blockDim.x + threadIdx.x;
  if (i < n) v[i] = 0.0f;
}

// One Sinkhorn half-iteration:
//   out[r] = -logsumexp_c( logk(r,c) + in_masked[c] )
// logk(r,c) = -d2/eps if d2 < 0.04 else -inf. Masked columns hold -1e30 so
// their exp terms are exactly 0. Column data lives in LDS; the running max
// is pre-subtracted into the LDS vector; inner reads are lane-uniform b128
// broadcasts; 4 independent partial sums break the v_exp latency chain.
__global__ void __launch_bounds__(64) sinkhorn_half(
    const float* __restrict__ row_locs,    // (B, 4096, 2) output side
    const float* __restrict__ col_locs,    // (B, 4096, 2) reduction side
    const uint8_t* __restrict__ col_mask,  // (B, 4096) mask applied to invec
    const float* __restrict__ invec,       // (B, 4096)
    float* __restrict__ outvec,            // (B, 4096)
    const uint8_t* __restrict__ out_mask)  // nullptr, or: out = mask? val : 0
{
  __shared__ __align__(16) float2 cl[S_];
  __shared__ __align__(16) float  cv[S_];
  __shared__ float sred[64];
  const int b   = blockIdx.y;
  const int tid = threadIdx.x;
  const float EINV = einv();

  for (int i = tid; i < S_; i += 64) {
    cl[i] = ((const float2*)col_locs)[b * S_ + i];
    cv[i] = col_mask[b * S_ + i] ? invec[b * S_ + i] : BIGNEG;
  }
  __syncthreads();

  // block-wide max of masked column vector for logsumexp stabilization
  float m = BIGNEG;
  for (int i = tid; i < S_; i += 64) m = fmaxf(m, cv[i]);
  sred[tid] = m;
  __syncthreads();
  for (int off = 32; off > 0; off >>= 1) {
    if (tid < off) sred[tid] = fmaxf(sred[tid], sred[tid + off]);
    __syncthreads();
  }
  const float cmax = sred[0];
  // fold -cmax into the LDS vector once (saves a VALU op * 4096 per row)
  for (int i = tid; i < S_; i += 64) cv[i] -= cmax;
  __syncthreads();

  const int r = blockIdx.x * 64 + tid;
  const float2 rl = ((const float2*)row_locs)[b * S_ + r];
  float sum0 = 0.0f, sum1 = 0.0f, sum2 = 0.0f, sum3 = 0.0f;
  for (int s = 0; s < S_; s += 4) {
    const float4 c01 = ((const float4*)cl)[(s >> 1) + 0];  // x0 y0 x1 y1
    const float4 c23 = ((const float4*)cl)[(s >> 1) + 1];  // x2 y2 x3 y3
    const float4 vv  = *(const float4*)(cv + s);
    float dx0 = rl.x - c01.x, dy0 = rl.y - c01.y;
    float dx1 = rl.x - c01.z, dy1 = rl.y - c01.w;
    float dx2 = rl.x - c23.x, dy2 = rl.y - c23.y;
    float dx3 = rl.x - c23.z, dy3 = rl.y - c23.w;
    float d0 = fmaf(dx0, dx0, dy0 * dy0);
    float d1 = fmaf(dx1, dx1, dy1 * dy1);
    float d2 = fmaf(dx2, dx2, dy2 * dy2);
    float d3 = fmaf(dx3, dx3, dy3 * dy3);
    float t0 = (d0 < T2F) ? __expf(fmaf(-d0, EINV, vv.x)) : 0.0f;
    float t1 = (d1 < T2F) ? __expf(fmaf(-d1, EINV, vv.y)) : 0.0f;
    float t2 = (d2 < T2F) ? __expf(fmaf(-d2, EINV, vv.z)) : 0.0f;
    float t3 = (d3 < T2F) ? __expf(fmaf(-d3, EINV, vv.w)) : 0.0f;
    sum0 += t0; sum1 += t1; sum2 += t2; sum3 += t3;
  }
  float sum = (sum0 + sum1) + (sum2 + sum3);
  float val = -(cmax + __logf(fmaxf(sum, 1e-37f)));
  if (out_mask) val = out_mask[b * S_ + r] ? val : 0.0f;
  outvec[b * S_ + r] = val;
}

// stage a 32(src) x 64(feat) fp32 tile into LDS transposed as f16:
// dst[n][k]; rows padded to 40 halves so per-lane 16B ds_load segments are
// bank-conflict free. 512 float4s / 128 threads = 4 each, fully unrolled.
__device__ __forceinline__ void stage_tile(const float* __restrict__ gtile,
                                           _Float16 (*__restrict__ dst)[40],
                                           int tid) {
#pragma unroll
  for (int j = 0; j < 4; ++j) {
    int idx = tid + j * 128;        // 0..511
    int k   = idx >> 4;             // 0..31
    int n4  = (idx & 15) << 2;      // 0,4,...,60
    const float4 f = *(const float4*)(gtile + k * D_ + n4);
    dst[n4 + 0][k] = (_Float16)f.x;
    dst[n4 + 1][k] = (_Float16)f.y;
    dst[n4 + 2][k] = (_Float16)f.z;
    dst[n4 + 3][k] = (_Float16)f.w;
  }
}

// Fused attn + einsum epilogue. Block = 128 threads = 4 waves; each wave owns
// a 16(target) x 64(feature) output tile, K over 4096 sources in steps of 32
// via v_wmma_f32_16x16x32_f16 (attn in [0,1] -> f16 safe, f32 accumulate).
// B tiles are double-buffered in LDS: stage s0+32 while computing s0.
__global__ void __launch_bounds__(128) ot_apply_wmma(
    const float* __restrict__ feats,   // (B, S, 64)
    const float* __restrict__ slocs,   // (B, S, 2)
    const float* __restrict__ tlocs,   // (B, T, 2)
    const uint8_t* __restrict__ smask, // (B, S)
    const uint8_t* __restrict__ tmask, // (B, T)
    const float* __restrict__ u,       // (B, T)
    const float* __restrict__ v,       // (B, S)
    float* __restrict__ out)           // (B, T, 64)
{
  __shared__ __align__(16) float2 sl[S_];
  __shared__ __align__(16) float  sv[S_];
  __shared__ __align__(16) _Float16 bt[2][D_][40];

  const int b    = blockIdx.y;
  const int tid  = threadIdx.x;
  const int lane = tid & 31;
  const int wave = tid >> 5;
  const int mrow = lane & 15;  // M within tile / B column
  const int hi   = lane >> 4;  // lane-half selects K subset
  const float EINV = einv();

  for (int i = tid; i < S_; i += 128) {
    sl[i] = ((const float2*)slocs)[b * S_ + i];
    sv[i] = smask[b * S_ + i] ? v[b * S_ + i] : BIGNEG;
  }

  const int t0 = blockIdx.x * 64 + wave * 16;
  const int t  = t0 + mrow;
  const float2 tl   = ((const float2*)tlocs)[b * T_ + t];
  const float  ureg = u[b * T_ + t];
  const float* gbase = feats + (size_t)b * S_ * D_;

  v8f acc[4] = {};

  stage_tile(gbase, bt[0], tid);
  __syncthreads();  // also covers sl/sv

  for (int s0 = 0; s0 < S_; s0 += 32) {
    const int p = (s0 >> 5) & 1;
    if (s0 + 32 < S_) {
      stage_tile(gbase + (size_t)(s0 + 32) * D_, bt[p ^ 1], tid);
      if (s0 + 64 < S_)  // speculative prefetch of the tile after next
        __builtin_prefetch(gbase + (size_t)(s0 + 64) * D_ + tid * 16, 0, 0);
    }

    // A fragment: 16x32 f16 attn tile per ISA layout; per lane the 16
    // K-values form two runs of 8 consecutive sources -> float4 LDS reads.
    //   lanes 0-15: e=0..7 -> K=0..7,  e=8..15 -> K=16..23
    //   lanes16-31: e=0..7 -> K=8..15, e=8..15 -> K=24..31
    v16h a;
#pragma unroll
    for (int half = 0; half < 2; ++half) {
      const int sA = s0 + half * 16 + hi * 8;  // multiple of 8
      const float4 p0 = ((const float4*)sl)[(sA >> 1) + 0];
      const float4 p1 = ((const float4*)sl)[(sA >> 1) + 1];
      const float4 p2 = ((const float4*)sl)[(sA >> 1) + 2];
      const float4 p3 = ((const float4*)sl)[(sA >> 1) + 3];
      const float4 v0 = *(const float4*)(sv + sA);
      const float4 v1 = *(const float4*)(sv + sA + 4);
      const float xs[8] = {p0.x, p0.z, p1.x, p1.z, p2.x, p2.z, p3.x, p3.z};
      const float ys[8] = {p0.y, p0.w, p1.y, p1.w, p2.y, p2.w, p3.y, p3.w};
      const float vs[8] = {v0.x, v0.y, v0.z, v0.w, v1.x, v1.y, v1.z, v1.w};
#pragma unroll
      for (int j = 0; j < 8; ++j) {
        float dx = tl.x - xs[j];
        float dy = tl.y - ys[j];
        float d2 = fmaf(dx, dx, dy * dy);
        float arg = fmaf(-d2, EINV, ureg + vs[j]);
        float av = (d2 < T2F) ? __expf(arg) : 0.0f;
        a[half * 8 + j] = (_Float16)av;
      }
    }

    // B fragments: 32x16 f16; lane column = mrow, K = hi*8 + e, e>=8 -> 16 + hi*8 + (e-8)
    const int kb = hi << 3;
#pragma unroll
    for (int nt = 0; nt < 4; ++nt) {
      v8h blo = *(const v8h*)&bt[p][nt * 16 + mrow][kb];
      v8h bhi = *(const v8h*)&bt[p][nt * 16 + mrow][kb + 16];
      v16h bf;
#pragma unroll
      for (int e = 0; e < 8; ++e) { bf[e] = blo[e]; bf[e + 8] = bhi[e]; }
      acc[nt] = __builtin_amdgcn_wmma_f32_16x16x32_f16(
          false, a, false, bf, (short)0, acc[nt], false, false);
    }
    __syncthreads();  // staging of bt[p^1] done; compute on bt[p] done
  }

  // C/D layout: VGPR r -> M = r + hi*8, N = lane&15. Rows with no valid
  // source are exactly 0 already (has_source implicit); apply target mask.
  float rmul[8];
#pragma unroll
  for (int r = 0; r < 8; ++r)
    rmul[r] = tmask[b * T_ + t0 + r + hi * 8] ? 1.0f : 0.0f;
#pragma unroll
  for (int nt = 0; nt < 4; ++nt) {
#pragma unroll
    for (int r = 0; r < 8; ++r) {
      int tr = t0 + r + hi * 8;
      out[((size_t)b * T_ + tr) * D_ + nt * 16 + mrow] = acc[nt][r] * rmul[r];
    }
  }
}

extern "C" void kernel_launch(void* const* d_in, const int* in_sizes, int n_in,
                              void* d_out, int out_size, void* d_ws, size_t ws_size,
                              hipStream_t stream) {
  (void)in_sizes; (void)n_in; (void)out_size; (void)ws_size;
  const float*   feats = (const float*)d_in[0];    // (4,4096,64) f32
  const float*   slocs = (const float*)d_in[1];    // (4,4096,2)  f32
  const float*   tlocs = (const float*)d_in[2];    // (4,4096,2)  f32
  const uint8_t* smask = (const uint8_t*)d_in[3];  // (4,4096) bool
  const uint8_t* tmask = (const uint8_t*)d_in[4];  // (4,4096) bool
  float* out = (float*)d_out;

  // workspace: u (B*T) then v (B*S) floats = 128 KB
  float* u = (float*)d_ws;
  float* v = u + B_ * T_;

  init_vec<<<dim3((B_ * S_ + 255) / 256), 256, 0, stream>>>(v, B_ * S_);

  for (int it = 0; it < NITER; ++it) {
    // u-update: rows = targets, cols = sources (v masked by smask)
    sinkhorn_half<<<dim3(T_ / 64, B_), 64, 0, stream>>>(
        tlocs, slocs, smask, v, u, nullptr);
    // v-update: rows = sources, cols = targets (u masked by tmask); v zeroed
    // where source invalid, matching jnp.where(src_mask, v, 0).
    sinkhorn_half<<<dim3(S_ / 64, B_), 64, 0, stream>>>(
        slocs, tlocs, tmask, u, v, smask);
  }

  ot_apply_wmma<<<dim3(T_ / 64, B_), 128, 0, stream>>>(
      feats, slocs, tlocs, smask, tmask, u, v, out);
}
